// EMA_VectorQuantizer_75230647157547
// MI455X (gfx1250) — matrix operand order repaired and checked by reading, not compiled, not run on hardware
//
#include <hip/hip_runtime.h>
#include <stdint.h>

#define ND 512
#define NN 65536
#define KK 8192
#define NTILES (KK / 16)

typedef __attribute__((ext_vector_type(16))) _Float16 v16h;
typedef __attribute__((ext_vector_type(8)))  _Float16 v8h;
typedef __attribute__((ext_vector_type(8)))  float    v8f;

union AFrag { v16h v; struct { v8h lo, hi; } p; };

// CDNA5 async global->LDS copy (ASYNCcnt-tracked, no VGPR staging).
// The instruction offset is added to both the LDS and the global address.
#define ASYNC_LD_16B(lds, gptr, imm)                                       \
  asm volatile("global_load_async_to_lds_b128 %0, %1, off offset:" #imm   \
               :: "v"(lds), "v"(gptr) : "memory")

__device__ __forceinline__ void async_stage64(uint32_t lds, const _Float16* g) {
  ASYNC_LD_16B(lds, g, 0);
  ASYNC_LD_16B(lds, g, 16);
  ASYNC_LD_16B(lds, g, 32);
  ASYNC_LD_16B(lds, g, 48);
}

__device__ __forceinline__ void wait_async0() {
  asm volatile("s_wait_asynccnt 0x0" ::: "memory");
}

__device__ __forceinline__ float warp_sum32(float x) {
#pragma unroll
  for (int m = 16; m >= 1; m >>= 1) x += __shfl_xor(x, m, 32);
  return x;
}

// Normalize rows of src [rows x 512] (F.normalize semantics) -> f16 dst.
// Optionally writes the post-normalization squared norm (== ||cb_k||^2 term).
__global__ void vq_row_normalize(const float* __restrict__ src,
                                 _Float16* __restrict__ dst,
                                 float* __restrict__ n2out) {
  const int row  = blockIdx.x * 8 + (threadIdx.x >> 5);
  const int lane = threadIdx.x & 31;
  const float* s = src + (size_t)row * ND;
  float v[16];
  float acc = 0.f;
#pragma unroll
  for (int i = 0; i < 16; ++i) {
    v[i] = s[lane + 32 * i];
    acc += v[i] * v[i];
  }
  acc = warp_sum32(acc);
  const float inv = 1.0f / fmaxf(sqrtf(acc), 1e-12f);
  _Float16* d = dst + (size_t)row * ND;
  float c2 = 0.f;
#pragma unroll
  for (int i = 0; i < 16; ++i) {
    const float q = v[i] * inv;
    c2 += q * q;
    d[lane + 32 * i] = (_Float16)q;
  }
  if (n2out) {
    c2 = warp_sum32(c2);
    if (lane == 0) n2out[row] = c2;
  }
}

// Main kernel: 256 threads handle 128 z rows (16/wave, A pinned in VGPRs).
// All 8192 codes stream through double-buffered LDS tiles filled with
// global_load_async_to_lds_b128; dots via v_wmma_f32_16x16x32_f16 on two
// independent accumulator chains; per-lane running argmax of (2*dot - ||cb||^2).
__global__ __launch_bounds__(256, 1)
void vq_argmin_wmma(const _Float16* __restrict__ zn,
                    const _Float16* __restrict__ cb,
                    const float* __restrict__ cn2,
                    int* __restrict__ idx_out) {
  __shared__ _Float16 sb[2][16][520];   // 520-half stride: 260 dwords -> conflict-free b128 reads

  const int tid  = threadIdx.x;
  const int lane = tid & 31;
  const int wave = tid >> 5;
  const int rw   = blockIdx.x * 128 + wave * 16;

  // ---- A fragments: 16 rows x 512 K, ISA 16-bit A 16x32 layout ----
  AFrag a[16];
  {
    const int m = rw + (lane & 15);
    const _Float16* ap = zn + (size_t)m * ND + ((lane >> 4) << 3);
#pragma unroll
    for (int kt = 0; kt < 16; ++kt) {
      a[kt].p.lo = *(const v8h*)(ap + kt * 32);        // K = base .. base+7
      a[kt].p.hi = *(const v8h*)(ap + kt * 32 + 16);   // K = base+16 .. base+23
    }
  }

  // ---- async staging: each thread owns a 64B chunk of the 16KB tile ----
  const int srow = tid >> 4;   // code row in tile (0..15)
  const int sseg = tid & 15;   // 32-half segment  (0..15)
  const _Float16* gB = cb + (size_t)srow * ND + sseg * 32;
  const uint32_t lds0 = (uint32_t)(size_t)&sb[0][srow][sseg * 32];  // low 32 bits = LDS offset
  const uint32_t lds1 = (uint32_t)(size_t)&sb[1][srow][sseg * 32];

  async_stage64(lds0, gB);          // tile 0 -> buf 0
  wait_async0();
  __syncthreads();

  float bestv[8];
  int   besti[8];
#pragma unroll
  for (int r = 0; r < 8; ++r) { bestv[r] = -3.4e38f; besti[r] = 0; }

  const int ncol = lane & 15;

  for (int t = 0; t < NTILES; ++t) {
    const int nb = t & 1;

    // start filling the other buffer (its last readers passed the previous barrier)
    if (t + 1 < NTILES)
      async_stage64(nb ? lds0 : lds1, gB + (size_t)(t + 1) * 16 * ND);

    // ---- 16 WMMAs over D=512; B frags pipelined 2 deep from LDS ----
    const _Float16* lb = &sb[nb][ncol][(lane >> 4) << 3];
    AFrag b[3];
    b[0].p.lo = *(const v8h*)(lb);      b[0].p.hi = *(const v8h*)(lb + 16);
    b[1].p.lo = *(const v8h*)(lb + 32); b[1].p.hi = *(const v8h*)(lb + 48);
    v8f c0 = {}, c1 = {};
#pragma unroll
    for (int kt = 0; kt < 16; ++kt) {
      if (kt + 2 < 16) {
        const _Float16* lp = lb + (kt + 2) * 32;
        b[(kt + 2) % 3].p.lo = *(const v8h*)lp;
        b[(kt + 2) % 3].p.hi = *(const v8h*)(lp + 16);
      }
      if (kt & 1)
        c1 = __builtin_amdgcn_wmma_f32_16x16x32_f16(false, a[kt].v, false, b[kt % 3].v,
                                                    (short)0, c1, false, false);
      else
        c0 = __builtin_amdgcn_wmma_f32_16x16x32_f16(false, a[kt].v, false, b[kt % 3].v,
                                                    (short)0, c0, false, false);
    }
    const v8f c = c0 + c1;

    const int   n_id = t * 16 + ncol;
    const float c2v  = cn2[n_id];
#pragma unroll
    for (int r = 0; r < 8; ++r) {
      const float s = 2.0f * c[r] - c2v;   // maximize  <=>  argmin distance
      if (s > bestv[r]) { bestv[r] = s; besti[r] = n_id; }
    }

    wait_async0();      // async fill of next buffer complete (barrier doesn't cover ASYNCcnt)
    __syncthreads();
  }

  // ---- cross-lane argmax within each 16-lane half (C layout: lane = col) ----
#pragma unroll
  for (int r = 0; r < 8; ++r) {
    float v  = bestv[r];
    int   bi = besti[r];
#pragma unroll
    for (int m = 8; m >= 1; m >>= 1) {
      const float ov = __shfl_xor(v, m, 32);
      const int   oi = __shfl_xor(bi, m, 32);
      if (ov > v || (ov == v && oi < bi)) { v = ov; bi = oi; }  // argmin-first ties
    }
    if ((lane & 15) == r)
      idx_out[rw + r + ((lane >> 4) << 3)] = bi;   // rows r (lanes 0-15), r+8 (lanes 16-31)
  }
}

// Exact f32 finalize: re-normalize z row and gathered codebook row, project.
__global__ void vq_finalize(const float* __restrict__ z,
                            const float* __restrict__ emb,
                            const int* __restrict__ idx,
                            float* __restrict__ out_proj,
                            float* __restrict__ out_idx) {
  const int row  = blockIdx.x * 8 + (threadIdx.x >> 5);
  const int lane = threadIdx.x & 31;
  const int ci   = idx[row];
  const float* zr = z   + (size_t)row * ND;
  const float* cr = emb + (size_t)ci  * ND;
  float zv[16], cv[16];
  float zs = 0.f, cs = 0.f;
#pragma unroll
  for (int i = 0; i < 16; ++i) {
    zv[i] = zr[lane + 32 * i]; zs += zv[i] * zv[i];
    cv[i] = cr[lane + 32 * i]; cs += cv[i] * cv[i];
  }
  zs = warp_sum32(zs);
  cs = warp_sum32(cs);
  const float zi   = 1.0f / fmaxf(sqrtf(zs), 1e-12f);
  const float cinv = 1.0f / fmaxf(sqrtf(cs), 1e-12f);
  float dot = 0.f;
#pragma unroll
  for (int i = 0; i < 16; ++i) dot += (zv[i] * zi) * (cv[i] * cinv);
  dot = warp_sum32(dot);
  const float scale = dot * cinv;           // (zn . z_q) * z_q,  z_q = cv*cinv
  float* op = out_proj + (size_t)row * ND;
#pragma unroll
  for (int i = 0; i < 16; ++i) op[lane + 32 * i] = scale * cv[i];
  if (lane == 0) out_idx[row] = (float)ci;
}

extern "C" void kernel_launch(void* const* d_in, const int* in_sizes, int n_in,
                              void* d_out, int out_size, void* d_ws, size_t ws_size,
                              hipStream_t stream) {
  const float* z   = (const float*)d_in[0];   // [N, D] f32
  const float* emb = (const float*)d_in[1];   // [K, D] f32

  char* ws = (char*)d_ws;
  const size_t cb_bytes = (size_t)KK * ND * sizeof(_Float16);   // 8 MB
  const size_t zn_bytes = (size_t)NN * ND * sizeof(_Float16);   // 64 MB
  _Float16* cb16 = (_Float16*)(ws);
  _Float16* zn16 = (_Float16*)(ws + cb_bytes);
  float*    cn2  = (float*)   (ws + cb_bytes + zn_bytes);
  int*      idx  = (int*)     (ws + cb_bytes + zn_bytes + (size_t)KK * sizeof(float));

  float* out_proj = (float*)d_out;                       // [N, D]
  float* out_idx  = (float*)d_out + (size_t)NN * ND;     // [N]

  vq_row_normalize<<<KK / 8, 256, 0, stream>>>(emb, cb16, cn2);
  vq_row_normalize<<<NN / 8, 256, 0, stream>>>(z, zn16, nullptr);
  vq_argmin_wmma<<<NN / 128, 256, 0, stream>>>(zn16, cb16, cn2, idx);
  vq_finalize<<<NN / 8, 256, 0, stream>>>(z, emb, idx, out_proj, out_idx);
}
